// _IPEXGatedMLPMOEXPU_41832981463135
// MI455X (gfx1250) — matrix operand order repaired and checked by reading, not compiled, not run on hardware
//
#include <hip/hip_runtime.h>
#include <hip/hip_bf16.h>

// ---------------- problem constants (from reference) ----------------
constexpr int kE = 8;      // experts
constexpr int kH = 2048;   // hidden
constexpr int kI = 5632;   // intermediate
constexpr int kT = 4096;   // tokens

// ---------------- vector types for WMMA ----------------
typedef __attribute__((ext_vector_type(16))) __bf16 v16bf;
typedef __attribute__((ext_vector_type(8)))  __bf16 bf16x8;
typedef __attribute__((ext_vector_type(4)))  __bf16 bf16x4;
typedef __attribute__((ext_vector_type(8)))  float  v8f;

__device__ __forceinline__ float silu_f(float x) {
    // x * 1/(1+e^-x) with fast v_rcp_f32 (no IEEE divide sequence)
    return x * __builtin_amdgcn_rcpf(1.0f + __expf(-x));
}

__device__ __forceinline__ void atomic_add_f32(float* p, float v) {
    __hip_atomic_fetch_add(p, v, __ATOMIC_RELAXED, __HIP_MEMORY_SCOPE_AGENT);
}

// =====================================================================
// 1) Routing: softmax over E logits, top-k, optional renormalize.
// =====================================================================
__global__ __launch_bounds__(256)
void route_topk(const float* __restrict__ logits,
                int* __restrict__ tids, float* __restrict__ tws,
                const int* __restrict__ pk, const int* __restrict__ prn) {
    int t = blockIdx.x * 256 + threadIdx.x;
    if (t >= kT) return;
    float p[kE];
    float mx = -3.4e38f;
#pragma unroll
    for (int e = 0; e < kE; ++e) { p[e] = logits[t * kE + e]; mx = fmaxf(mx, p[e]); }
    float s = 0.f;
#pragma unroll
    for (int e = 0; e < kE; ++e) { p[e] = __expf(p[e] - mx); s += p[e]; }
    float inv = 1.0f / s;
#pragma unroll
    for (int e = 0; e < kE; ++e) p[e] *= inv;

    int K = pk[0]; if (K > kE) K = kE; if (K < 1) K = 1;
    int   iv[kE];
    float wv[kE];
    float wsum = 0.f;
    for (int kk = 0; kk < K; ++kk) {
        int am = 0; float bm = -1.f;
#pragma unroll
        for (int e = 0; e < kE; ++e) { if (p[e] > bm) { bm = p[e]; am = e; } }
        iv[kk] = am; wv[kk] = bm; wsum += bm; p[am] = -2.f;
    }
    float scale = prn[0] ? (1.0f / wsum) : 1.0f;
    for (int kk = 0; kk < K; ++kk) {
        tids[t * K + kk] = iv[kk];
        tws[t * K + kk]  = wv[kk] * scale;
    }
}

// =====================================================================
// 2) Build compacted per-expert slot lists (single block).
// =====================================================================
__global__ __launch_bounds__(256)
void build_lists(const int* __restrict__ tids, const float* __restrict__ tws,
                 const int* __restrict__ pk,
                 int* __restrict__ counts, int* __restrict__ offsets,
                 int* __restrict__ ids, float* __restrict__ coefs) {
    __shared__ int scnt[kE];
    __shared__ int soff[kE];
    __shared__ int scur[kE];
    int tx = threadIdx.x;
    if (tx < kE) scnt[tx] = 0;
    __syncthreads();
    int K = pk[0]; if (K > kE) K = kE; if (K < 1) K = 1;
    int N = kT * K;
    for (int i = tx; i < N; i += 256) atomicAdd(&scnt[tids[i]], 1);
    __syncthreads();
    if (tx == 0) {
        int run = 0;
        for (int e = 0; e < kE; ++e) { soff[e] = run; run += scnt[e]; }
    }
    __syncthreads();
    if (tx < kE) { scur[tx] = 0; counts[tx] = scnt[tx]; offsets[tx] = soff[tx]; }
    __syncthreads();
    for (int i = tx; i < N; i += 256) {
        int e = tids[i];
        int pos = atomicAdd(&scur[e], 1);
        int slot = soff[e] + pos;
        ids[slot]   = i / K;     // token
        coefs[slot] = tws[i];    // routing weight (renormalized)
    }
}

// =====================================================================
// 3) Convert hidden_states fp32 -> bf16 (4 elems / thread).
// =====================================================================
__global__ __launch_bounds__(256)
void cvt_bf16(const float* __restrict__ x, __bf16* __restrict__ y, int n4) {
    int i = blockIdx.x * 256 + threadIdx.x;
    if (i >= n4) return;
    const float4 f = reinterpret_cast<const float4*>(x)[i];
    bf16x4 o;
    o[0] = (__bf16)f.x; o[1] = (__bf16)f.y; o[2] = (__bf16)f.z; o[3] = (__bf16)f.w;
    *reinterpret_cast<bf16x4*>(y + (size_t)i * 4) = o;
}

// =====================================================================
// 4) Repack weights: [E][R][C] fp32 row-major -> [E][C][R] bf16
//    (column-panel layout: K contiguous per output column).
//    32x32 tiles staged through LDS; coalesced read and write.
// =====================================================================
__global__ __launch_bounds__(256)
void tr_cvt(const float* __restrict__ src, __bf16* __restrict__ dst,
            int R, int C) {
    __shared__ __bf16 tile[32][33];
    const size_t eoff = (size_t)blockIdx.z * R * C;
    src += eoff;
    dst += eoff;
    const int r0 = blockIdx.x * 32;
    const int c0 = blockIdx.y * 32;
    const int tx = threadIdx.x & 31;
    const int ty = threadIdx.x >> 5;   // 0..7
#pragma unroll
    for (int i = ty; i < 32; i += 8)
        tile[i][tx] = (__bf16)src[(size_t)(r0 + i) * C + (c0 + tx)];
    __syncthreads();
#pragma unroll
    for (int i = ty; i < 32; i += 8)
        dst[(size_t)(c0 + i) * R + (r0 + tx)] = tile[tx][i];
}

// =====================================================================
// WMMA fragment helpers (16x16x32 bf16, wave32).
// A (16x32): lane holds row lane%16; elem j -> K = 16*(lane/16)+j
// B (32x16): lane holds col lane%16; elem j -> K = 16*(lane/16)+j
// D (16x16): VGPR v: row = v + 8*(lane/16), col = lane%16
// =====================================================================
__device__ __forceinline__ v16bf load_a_frag(const __bf16* __restrict__ arow,
                                             int k, int kh) {
    bf16x8 c0 = *reinterpret_cast<const bf16x8*>(arow + k + kh * 16);
    bf16x8 c1 = *reinterpret_cast<const bf16x8*>(arow + k + kh * 16 + 8);
    v16bf a;
#pragma unroll
    for (int j = 0; j < 8; ++j) { a[j] = c0[j]; a[8 + j] = c1[j]; }
    return a;
}

// Column-panel bf16 weights: 16 contiguous K values -> one 32B load.
__device__ __forceinline__ v16bf load_b_frag(const __bf16* __restrict__ col,
                                             int k, int kh) {
    return *reinterpret_cast<const v16bf*>(col + k + kh * 16);
}

#define WMMA_BF16(ACC, A, B) \
    (ACC) = __builtin_amdgcn_wmma_f32_16x16x32_bf16(false, (A), false, (B), \
                                                    (short)0, (ACC), false, false)

// =====================================================================
// 5) GEMM1: h[slot, n] = silu(X@Wg) * (X@Wu) * coef   (bf16 out)
//    Block: 64 rows x 128 cols. 8 waves as 2x4; each wave owns a
//    32x32 strip = 2 m-subtiles x 2 n-subtiles, gate+up accumulators.
//    Per 32-K step: 2 A frags + 4 B frags -> 8 WMMAs.
// =====================================================================
__global__ __launch_bounds__(256)
void moe_gemm1(const __bf16* __restrict__ Xbf, const __bf16* __restrict__ W13t,
               const int* __restrict__ counts, const int* __restrict__ offsets,
               const int* __restrict__ ids, const float* __restrict__ coefs,
               __bf16* __restrict__ h_ws) {
    const int e      = blockIdx.x >> 6;
    const int mtile  = blockIdx.x & 63;
    const int cnt    = counts[e];
    const int mstart = mtile * 64;
    if (mstart >= cnt) return;
    const int off  = offsets[e];
    const int lane = threadIdx.x & 31;
    const int wv   = threadIdx.x >> 5;
    const int wr   = wv >> 2;           // 0..1 (32-row strip)
    const int wc   = wv & 3;            // 0..3 (32-col strip)
    const int n0   = blockIdx.y * 128 + wc * 32;
    const int kh   = lane >> 4;
    const int lc   = lane & 15;

    // Two A rows per lane (m-subtile 0 and 1), clamped for ragged tail
    const int r0  = mstart + wr * 32 + lc;
    const int r1  = r0 + 16;
    const int r0c = r0 < cnt ? r0 : cnt - 1;
    const int r1c = r1 < cnt ? r1 : cnt - 1;
    const __bf16* arow0 = Xbf + (size_t)ids[off + r0c] * kH;
    const __bf16* arow1 = Xbf + (size_t)ids[off + r1c] * kH;

    // W13t layout: [E][2I][H] bf16, column-panel (K=H contiguous)
    const size_t eoff = (size_t)e * (2 * kI) * kH;
    const __bf16* bg0 = W13t + eoff + (size_t)(n0 + lc) * kH;
    const __bf16* bg1 = bg0 + (size_t)16 * kH;
    const __bf16* bu0 = W13t + eoff + (size_t)(kI + n0 + lc) * kH;
    const __bf16* bu1 = bu0 + (size_t)16 * kH;

    v8f ag00 = {}, ag01 = {}, ag10 = {}, ag11 = {};
    v8f au00 = {}, au01 = {}, au10 = {}, au11 = {};
    for (int k = 0; k < kH; k += 32) {
        if ((k & 255) == 0) {   // speculative prefetch of upcoming panels
            __builtin_prefetch(arow0 + k + 256, 0, 0);
            __builtin_prefetch(arow1 + k + 256, 0, 0);
            __builtin_prefetch(bg0 + k + 256, 0, 0);
            __builtin_prefetch(bu0 + k + 256, 0, 0);
        }
        v16bf a0 = load_a_frag(arow0, k, kh);
        v16bf a1 = load_a_frag(arow1, k, kh);
        v16bf g0 = load_b_frag(bg0, k, kh);
        v16bf g1 = load_b_frag(bg1, k, kh);
        v16bf u0 = load_b_frag(bu0, k, kh);
        v16bf u1 = load_b_frag(bu1, k, kh);
        WMMA_BF16(ag00, a0, g0);
        WMMA_BF16(ag01, a0, g1);
        WMMA_BF16(au00, a0, u0);
        WMMA_BF16(au01, a0, u1);
        WMMA_BF16(ag10, a1, g0);
        WMMA_BF16(ag11, a1, g1);
        WMMA_BF16(au10, a1, u0);
        WMMA_BF16(au11, a1, u1);
    }

#pragma unroll
    for (int v = 0; v < 8; ++v) {
        const int rr0 = mstart + wr * 32 + (v + 8 * kh);
        if (rr0 < cnt) {
            const int slot = off + rr0;
            const float c  = coefs[slot];
            __bf16* hrow = h_ws + (size_t)slot * kI;
            hrow[n0 + lc]      = (__bf16)(silu_f(ag00[v]) * au00[v] * c);
            hrow[n0 + 16 + lc] = (__bf16)(silu_f(ag01[v]) * au01[v] * c);
        }
        const int rr1 = rr0 + 16;
        if (rr1 < cnt) {
            const int slot = off + rr1;
            const float c  = coefs[slot];
            __bf16* hrow = h_ws + (size_t)slot * kI;
            hrow[n0 + lc]      = (__bf16)(silu_f(ag10[v]) * au10[v] * c);
            hrow[n0 + 16 + lc] = (__bf16)(silu_f(ag11[v]) * au11[v] * c);
        }
    }
}

// =====================================================================
// 6) GEMM2: out[token, n] += h[slot, :] @ W2[e]   (coef already folded)
//    Same 2x2 subtile wave layout: 2 A + 2 B frags -> 4 WMMAs / K-step.
// =====================================================================
__global__ __launch_bounds__(256)
void moe_gemm2(const __bf16* __restrict__ h_ws, const __bf16* __restrict__ W2t,
               const int* __restrict__ counts, const int* __restrict__ offsets,
               const int* __restrict__ ids, float* __restrict__ out) {
    const int e      = blockIdx.x >> 6;
    const int mtile  = blockIdx.x & 63;
    const int cnt    = counts[e];
    const int mstart = mtile * 64;
    if (mstart >= cnt) return;
    const int off  = offsets[e];
    const int lane = threadIdx.x & 31;
    const int wv   = threadIdx.x >> 5;
    const int wr   = wv >> 2;
    const int wc   = wv & 3;
    const int n0   = blockIdx.y * 128 + wc * 32;
    const int kh   = lane >> 4;
    const int lc   = lane & 15;

    const int r0  = mstart + wr * 32 + lc;
    const int r1  = r0 + 16;
    const int r0c = r0 < cnt ? r0 : cnt - 1;
    const int r1c = r1 < cnt ? r1 : cnt - 1;
    const __bf16* arow0 = h_ws + (size_t)(off + r0c) * kI;
    const __bf16* arow1 = h_ws + (size_t)(off + r1c) * kI;

    // W2t layout: [E][H][I] bf16, column-panel (K=I contiguous)
    const size_t eoff = (size_t)e * kH * kI;
    const __bf16* b0 = W2t + eoff + (size_t)(n0 + lc) * kI;
    const __bf16* b1 = b0 + (size_t)16 * kI;

    v8f acc00 = {}, acc01 = {}, acc10 = {}, acc11 = {};
    for (int k = 0; k < kI; k += 32) {
        if ((k & 255) == 0) {
            __builtin_prefetch(arow0 + k + 256, 0, 0);
            __builtin_prefetch(arow1 + k + 256, 0, 0);
            __builtin_prefetch(b0 + k + 256, 0, 0);
        }
        v16bf a0 = load_a_frag(arow0, k, kh);
        v16bf a1 = load_a_frag(arow1, k, kh);
        v16bf w0 = load_b_frag(b0, k, kh);
        v16bf w1 = load_b_frag(b1, k, kh);
        WMMA_BF16(acc00, a0, w0);
        WMMA_BF16(acc01, a0, w1);
        WMMA_BF16(acc10, a1, w0);
        WMMA_BF16(acc11, a1, w1);
    }

#pragma unroll
    for (int v = 0; v < 8; ++v) {
        const int rr0 = mstart + wr * 32 + (v + 8 * kh);
        if (rr0 < cnt) {
            const int slot = off + rr0;
            float* orow = out + (size_t)ids[slot] * kH;
            atomic_add_f32(&orow[n0 + lc],      acc00[v]);
            atomic_add_f32(&orow[n0 + 16 + lc], acc01[v]);
        }
        const int rr1 = rr0 + 16;
        if (rr1 < cnt) {
            const int slot = off + rr1;
            float* orow = out + (size_t)ids[slot] * kH;
            atomic_add_f32(&orow[n0 + lc],      acc10[v]);
            atomic_add_f32(&orow[n0 + 16 + lc], acc11[v]);
        }
    }
}

// =====================================================================
// Launcher
// =====================================================================
extern "C" void kernel_launch(void* const* d_in, const int* in_sizes, int n_in,
                              void* d_out, int out_size, void* d_ws, size_t ws_size,
                              hipStream_t stream) {
    (void)in_sizes; (void)n_in; (void)ws_size;
    const float* hs     = (const float*)d_in[0];   // [T, H]
    const float* logits = (const float*)d_in[1];   // [T, E]
    const float* W13    = (const float*)d_in[2];   // [E, H, 2I]
    const float* W2     = (const float*)d_in[3];   // [E, I, H]
    const int*   top_k  = (const int*)d_in[4];
    const int*   renorm = (const int*)d_in[6];
    float* out = (float*)d_out;                    // [T, H] fp32

    // ---- workspace layout ----
    char* w = (char*)d_ws;
    size_t o = 0;
    auto alloc = [&](size_t bytes) -> char* {
        o = (o + 255) & ~(size_t)255;
        char* p = w + o;
        o += bytes;
        return p;
    };
    int*    counts  = (int*)alloc(kE * 4);
    int*    offsets = (int*)alloc(kE * 4);
    int*    topk_id = (int*)alloc((size_t)kT * 2 * 4);
    float*  topk_w  = (float*)alloc((size_t)kT * 2 * 4);
    int*    ids     = (int*)alloc((size_t)kT * 2 * 4);
    float*  coefs   = (float*)alloc((size_t)kT * 2 * 4);
    __bf16* Xbf     = (__bf16*)alloc((size_t)kT * kH * 2);              // 16 MB
    __bf16* h_ws    = (__bf16*)alloc((size_t)kT * 2 * kI * 2);          // ~92 MB
    __bf16* W13t    = (__bf16*)alloc((size_t)kE * 2 * kI * kH * 2);     // ~369 MB
    __bf16* W2t     = (__bf16*)alloc((size_t)kE * kI * kH * 2);         // ~184 MB

    // 1) routing
    route_topk<<<kT / 256, 256, 0, stream>>>(logits, topk_id, topk_w, top_k, renorm);
    // 2) compaction
    build_lists<<<1, 256, 0, stream>>>(topk_id, topk_w, top_k, counts, offsets, ids, coefs);
    // 3) activation convert
    cvt_bf16<<<(kT * kH / 4 + 255) / 256, 256, 0, stream>>>(hs, Xbf, kT * kH / 4);
    // 4) weight repack: fp32 row-major -> bf16 column-panel
    dim3 gw13(kH / 32, 2 * kI / 32, kE);
    tr_cvt<<<gw13, 256, 0, stream>>>(W13, W13t, kH, 2 * kI);
    dim3 gw2(kI / 32, kH / 32, kE);
    tr_cvt<<<gw2, 256, 0, stream>>>(W2, W2t, kI, kH);
    // 5) zero output (gemm2 accumulates)
    hipMemsetAsync(d_out, 0, (size_t)out_size * sizeof(float), stream);
    // 6) gemm1 + silu*mul*coef
    dim3 g1(kE * (kT / 64), kI / 128);
    moe_gemm1<<<g1, 256, 0, stream>>>(Xbf, W13t, counts, offsets, ids, coefs, h_ws);
    // 7) gemm2 + scatter-add
    dim3 g2(kE * (kT / 64), kH / 128);
    moe_gemm2<<<g2, 256, 0, stream>>>(h_ws, W2t, counts, offsets, ids, out);
}
